// Model_30064771072898
// MI455X (gfx1250) — matrix operand order repaired and checked by reading, not compiled, not run on hardware
//
#include <hip/hip_runtime.h>
#include <math.h>
#include <stdint.h>

// ---------------------------------------------------------------------------
// CDNA5 (gfx1250) implementation. Wave32, WMMA bf16 16x16x32 with f32 accum.
// Fragment layouts (C/D as in ISA 7.12.2; A/B k order is this code's own):
//   A (16x32 bf16):  lane l, elem i -> M = l&15, K = 16*(l>>4) + i
//   B (32x16 bf16):  lane l, elem i -> K = 16*(l>>4) + i, N = l&15
//   C/D (16x16 f32): lane l, vgpr j -> M = (l<16 ? j : j+8), N = l&15
// All staged fragment buffers give each lane a contiguous 32B load.
// k_flow stages its X tile through LDS with the Tensor Data Mover (TDM),
// using descriptor padding (4 DW per 256 DW) to de-conflict LDS banks.
// ---------------------------------------------------------------------------

typedef __attribute__((ext_vector_type(16))) __bf16 v16bf;
typedef __attribute__((ext_vector_type(8)))  float  v8f;
typedef __attribute__((ext_vector_type(4)))  unsigned int u32x4;
typedef __attribute__((ext_vector_type(8)))  int  i32x8;
typedef __attribute__((ext_vector_type(4)))  int  i32x4;

#ifndef __has_builtin
#define __has_builtin(x) 0
#endif
#if __has_builtin(__builtin_amdgcn_tensor_load_to_lds) && \
    __has_builtin(__builtin_amdgcn_s_wait_tensorcnt)
#define HAVE_TDM 1
#else
#define HAVE_TDM 0
#endif

#define B_   32
#define L_   16
#define O_   256
#define DM_  64
#define NROW (B_ * O_)   /* 8192 LSTM rows */

// workspace byte offsets
#define OFF_GEO    0u                       // 256*64 f32            (64 KB)
#define OFF_GBUF   65536u                   // 256 tiles *32*8 f32   (256 KB)
#define OFF_WI1    327680u                  // 32 frags *512 bf16    (32 KB)
#define OFF_WH     360448u                  // 32 frags *512 bf16    (32 KB)
#define OFF_WL     393216u                  // 8 frags *512 bf16     (8 KB)
#define OFF_STATS  401408u                  // 128 f32 (sum, sumsq)
#define OFF_MEAN   401920u                  // 64 f32
#define OFF_RSTD   402176u                  // 64 f32
#define OFF_SEQX   409600u                  // 8192*16*64 bf16       (16 MB)
#define OFF_HLAST  (409600u + 16777216u)    // 8192*64 f32           (2 MB)

// native bf16 convert (lowers to v_cvt_pk_bf16_f32-class ops on gfx1250)
static __device__ inline __bf16 tobf(float f) { return (__bf16)f; }

static __device__ inline v16bf cvt16f(const float* p) {
  v16bf r;
#pragma unroll
  for (int i = 0; i < 16; ++i) r[i] = tobf(p[i]);
  return r;
}

static __device__ inline v8f v8f_zero() {
  v8f z;
#pragma unroll
  for (int j = 0; j < 8; ++j) z[j] = 0.f;
  return z;
}

static __device__ inline v8f wmma_bf16(v16bf a, v16bf b, v8f c) {
  return __builtin_amdgcn_wmma_f32_16x16x32_bf16(false, a, false, b,
                                                 (short)0, c, false, false);
}

// fragment buffer: frag-major, lane-major, 16 bf16 (32B) per lane
static __device__ inline v16bf ldfragG(const char* base, int frag, int lane) {
  return *(const v16bf*)(base + (size_t)frag * 1024 + (size_t)lane * 32);
}

static __device__ inline float sigf(float x) { return 1.f / (1.f + __expf(-x)); }

// ---------------------------------------------------------------------------
__global__ void k_zero(char* ws) {
  float* stats = (float*)(ws + OFF_STATS);
  if (threadIdx.x < 128) stats[threadIdx.x] = 0.f;
}

// attn = softmax(-dis, axis=-1); geo = attn @ E   (one block, 256 threads)
__global__ void k_geo(const float* dis, const float* E, float* geo) {
  int r = threadIdx.x;
  const float* dr = dis + r * O_;
  float m = -3.4e38f;
  for (int j = 0; j < O_; ++j) m = fmaxf(m, -dr[j]);
  float acc[DM_];
#pragma unroll
  for (int e = 0; e < DM_; ++e) acc[e] = 0.f;
  float s = 0.f;
  for (int j = 0; j < O_; ++j) {
    float p = __expf(-dr[j] - m);
    s += p;
    const float* Ej = E + j * DM_;
#pragma unroll
    for (int e = 0; e < DM_; ++e) acc[e] += p * Ej[e];
  }
  float inv = 1.f / s;
  for (int e = 0; e < DM_; ++e) geo[r * DM_ + e] = acc[e] * inv;
}

// Pre-format Wi (flow half), Wh, Wl into B-fragment layout (bf16)
__global__ void k_prepw(const float* Wi, const float* Wh, const float* Wl,
                        char* ws) {
  __bf16* wi1 = (__bf16*)(ws + OFF_WI1);
  __bf16* wh  = (__bf16*)(ws + OFF_WH);
  __bf16* wl  = (__bf16*)(ws + OFF_WL);
  int tid = blockIdx.x * blockDim.x + threadIdx.x;
  int stride = gridDim.x * blockDim.x;
  for (int idx = tid; idx < 32 * 512; idx += stride) {
    int frag = idx >> 9, lane = (idx >> 4) & 31, i = idx & 15;
    int c = frag >> 4, n = frag & 15;
    // B frag elem (lane,i): K = c*32+16*(lane>>4)+i (feature), N = n*16+(lane&15) (gate)
    wi1[idx] = tobf(Wi[(n * 16 + (lane & 15)) * (2 * DM_) + (c * 32 + 16 * (lane >> 4) + i)]);  // cols 0..63
    wh[idx]  = tobf(Wh[(n * 16 + (lane & 15)) * DM_ + (c * 32 + 16 * (lane >> 4) + i)]);
  }
  for (int idx = tid; idx < 8 * 512; idx += stride) {
    int frag = idx >> 9, lane = (idx >> 4) & 31, i = idx & 15;
    int c = frag >> 2, n = frag & 3;
    wl[idx] = tobf(Wl[(n * 16 + (lane & 15)) * DM_ + (c * 32 + 16 * (lane >> 4) + i)]);
  }
}

// G[o,gate] = geo[o,:] @ Wi[:,64:128]^T + bi + bh, stored in C-fragment order.
__global__ void k_gtot(const float* Wi, const float* bi, const float* bh,
                       char* ws) {
  const float* geo = (const float*)(ws + OFF_GEO);
  float* gbuf = (float*)(ws + OFF_GBUF);
  int lane = threadIdx.x & 31;
  int w = blockIdx.x * 8 + (threadIdx.x >> 5);  // 0..255
  int ot = w >> 4, nt = w & 15;
  int g0 = nt * 16;
  float bsum = bi[g0 + (lane & 15)] + bh[g0 + (lane & 15)];
  v8f acc;
#pragma unroll
  for (int j = 0; j < 8; ++j) acc[j] = bsum;
#pragma unroll
  for (int c = 0; c < 2; ++c) {
    v16bf a = cvt16f(geo + (ot * 16 + (lane & 15)) * DM_ + c * 32 + 16 * (lane >> 4));
    v16bf b;
#pragma unroll
    for (int i = 0; i < 16; ++i)
      b[i] = tobf(Wi[(g0 + (lane & 15)) * (2 * DM_) + DM_ + c * 32 + 16 * (lane >> 4) + i]);
    acc = wmma_bf16(a, b, acc);
  }
#pragma unroll
  for (int j = 0; j < 8; ++j) gbuf[w * 256 + lane * 8 + j] = acc[j];
}

// feat_flow GEMM: (b,t,o-tile) wave -> 16x64 tile = X[b,t,o0:o0+16,:] @ E.
// A-operand (16x256 f32, contiguous 16KB) staged LDS via TDM with bank-depad.
#define TDM_RS   260                 /* padded LDS row stride in f32 */
#define TDM_SLOT 17408               /* per-wave LDS bytes (16*260*4 = 16640) */
__global__ void __launch_bounds__(128, 1) k_flow(const float* X, const float* E,
                                                 char* ws) {
  extern __shared__ char smem[];
  __bf16* seqx = (__bf16*)(ws + OFF_SEQX);
  int lane = threadIdx.x & 31;
  int widx = threadIdx.x >> 5;
  int w = blockIdx.x * 4 + widx;  // 0..8191
  int b = w >> 8, t = (w >> 4) & 15, o0 = (w & 15) * 16;
  const float* Xtile = X + (((size_t)b * L_ + t) * O_ + o0) * O_;  // 16KB slab

#if HAVE_TDM
  float* lds_tile = (float*)(smem + widx * TDM_SLOT);
  {
    // D# group 0: count=1, lds_addr, global_addr, type=2 (ISA 8.3)
    unsigned lds_off = (unsigned)(uintptr_t)lds_tile;  // low 32b = LDS offset
    unsigned long long ga = (unsigned long long)(uintptr_t)Xtile;
    u32x4 g0;
    g0[0] = 1u;                                  // count=1, user-mode
    g0[1] = lds_off;                             // lds_addr
    g0[2] = (unsigned)(ga & 0xffffffffu);        // global_addr[31:0]
    g0[3] = (unsigned)((ga >> 32) & 0x01ffffffu) // global_addr[56:32]
            | 0x80000000u;                       // type=2
    // D# group 1: data_size=4B, pad 4DW per 256DW, dims (ISA 8.4)
    i32x8 g1;
    g1[0] = (int)((2u << 16)        // data_size = 4 bytes
                  | (1u << 20)      // pad_enable
                  | (7u << 22)      // pad_interval = 256 DWORDs
                  | (3u << 25));    // pad_amount = 4 DWORDs
    g1[1] = (int)(4096u << 16);     // tensor_dim0[15:0] (=4096 elements)
    g1[2] = (int)(1u << 16);        // tensor_dim0 hi=0 | tensor_dim1 lo=1
    g1[3] = (int)(4096u << 16);     // tensor_dim1 hi=0 | tile_dim0=4096
    g1[4] = 1;                      // tile_dim1=1, tile_dim2=0
    g1[5] = 4096;                   // tensor_dim0_stride lo
    g1[6] = 0;
    g1[7] = 0;
    i32x4 z4 = {0, 0, 0, 0};
#if defined(__clang_major__) && __clang_major__ >= 23
    i32x8 z8 = {0, 0, 0, 0, 0, 0, 0, 0};
    __builtin_amdgcn_tensor_load_to_lds(g0, g1, z4, z4, z8, 0);
#else
    __builtin_amdgcn_tensor_load_to_lds(g0, g1, z4, z4, 0);
#endif
    __builtin_amdgcn_s_wait_tensorcnt(0);
  }
  const float* Asrc = lds_tile;
  const int RS = TDM_RS;
#else
  const float* Asrc = Xtile;
  const int RS = O_;
#endif

  v8f acc[4];
#pragma unroll
  for (int n = 0; n < 4; ++n) acc[n] = v8f_zero();
#pragma unroll
  for (int c = 0; c < 8; ++c) {
    v16bf a = cvt16f(Asrc + (size_t)(lane & 15) * RS + c * 32 + 16 * (lane >> 4));
#pragma unroll
    for (int n = 0; n < 4; ++n) {
      v16bf bb;
#pragma unroll
      for (int i = 0; i < 16; ++i)
        bb[i] = tobf(E[(c * 32 + 16 * (lane >> 4) + i) * DM_ + n * 16 + (lane & 15)]);
      acc[n] = wmma_bf16(a, bb, acc[n]);
    }
  }
#pragma unroll
  for (int n = 0; n < 4; ++n)
#pragma unroll
    for (int j = 0; j < 8; ++j) {
      int M = (lane < 16) ? j : j + 8;
      size_t row = (size_t)b * O_ + o0 + M;
      seqx[(row * L_ + t) * DM_ + n * 16 + (lane & 15)] = tobf(acc[n][j]);
    }
}

// LSTM: wave owns a 16-row batch tile for all 16 steps.
__global__ void __launch_bounds__(256, 1) k_lstm(char* ws) {
  extern __shared__ char smem[];
  const __bf16* seqx = (const __bf16*)(ws + OFF_SEQX);
  const float* gbuf = (const float*)(ws + OFF_GBUF);
  const char* wi1c = ws + OFF_WI1;
  const char* whc = ws + OFF_WH;
  float* hlast = (float*)(ws + OFF_HLAST);
  float* stats = (float*)(ws + OFF_STATS);

  int lane = threadIdx.x & 31;
  int widx = threadIdx.x >> 5;
  int tile = blockIdx.x * 8 + widx;  // 0..511
  int b = tile >> 4;
  int otile = tile & 15;
  int o0 = otile * 16;
  char* hscr = smem + widx * 2048;  // wave-private 16x64 bf16 scratch

  // pull weight fragment buffers toward this WGP (global_prefetch_b8)
  __builtin_prefetch(wi1c + (size_t)lane * 1024, 0, 1);
  __builtin_prefetch(whc + (size_t)lane * 1024, 0, 1);

  v8f cst[4];
#pragma unroll
  for (int n = 0; n < 4; ++n) cst[n] = v8f_zero();
  v16bf hf[2];
#pragma unroll
  for (int c = 0; c < 2; ++c)
#pragma unroll
    for (int i = 0; i < 16; ++i) hf[c][i] = tobf(0.f);
  float sh[4] = {0, 0, 0, 0}, sh2[4] = {0, 0, 0, 0};

  size_t rowbase = (size_t)b * O_ + o0;

  for (int t = 0; t < L_; ++t) {
    v16bf xf[2];
#pragma unroll
    for (int c = 0; c < 2; ++c) {
      size_t off = ((rowbase + (lane & 15)) * L_ + t) * DM_ + c * 32 + 16 * (lane >> 4);
      xf[c] = *(const v16bf*)(seqx + off);
    }
    v8f acc[16];
#pragma unroll
    for (int n = 0; n < 16; ++n) {
      acc[n] = *(const v8f*)(gbuf + ((otile * 16 + n) * 256 + lane * 8));
      acc[n] = wmma_bf16(xf[0], ldfragG(wi1c, n, lane), acc[n]);
      acc[n] = wmma_bf16(xf[1], ldfragG(wi1c, 16 + n, lane), acc[n]);
      acc[n] = wmma_bf16(hf[0], ldfragG(whc, n, lane), acc[n]);
      acc[n] = wmma_bf16(hf[1], ldfragG(whc, 16 + n, lane), acc[n]);
    }
    bool last = (t == L_ - 1);
#pragma unroll
    for (int n = 0; n < 4; ++n)
#pragma unroll
      for (int j = 0; j < 8; ++j) {
        float ig = sigf(acc[n][j]);
        float fg = sigf(acc[n + 4][j]);
        float gg = tanhf(acc[n + 8][j]);
        float og = sigf(acc[n + 12][j]);
        float cc = fg * cst[n][j] + ig * gg;
        cst[n][j] = cc;
        float hh = og * tanhf(cc);
        sh[n] += hh;
        sh2[n] += hh * hh;
        int M = (lane < 16) ? j : j + 8;
        ((__bf16*)hscr)[M * DM_ + n * 16 + (lane & 15)] = tobf(hh);
        if (last) hlast[(rowbase + M) * DM_ + n * 16 + (lane & 15)] = hh;
      }
    // gather h back as A fragments for the next step (wave-local LDS, in order)
#pragma unroll
    for (int c = 0; c < 2; ++c)
      hf[c] = *(const v16bf*)(hscr + ((lane & 15) * DM_ + c * 32 + 16 * (lane >> 4)) * 2);
  }
#pragma unroll
  for (int n = 0; n < 4; ++n) {
    int f = n * 16 + (lane & 15);
    atomicAdd(&stats[f], sh[n]);
    atomicAdd(&stats[64 + f], sh2[n]);
  }
}

__global__ void k_finalize(char* ws) {
  int f = threadIdx.x;  // 64
  float* stats = (float*)(ws + OFF_STATS);
  float cnt = (float)(NROW * L_);
  float mean = stats[f] / cnt;
  float var = stats[64 + f] / cnt - mean * mean;
  ((float*)(ws + OFF_MEAN))[f] = mean;
  ((float*)(ws + OFF_RSTD))[f] = rsqrtf(var + 1e-5f);
}

// Head: per batch b -> normalize v, lv = v@Wl^T + bl, out = lv @ v^T
__global__ void __launch_bounds__(256, 1) k_head(const float* gamma,
                                                 const float* beta,
                                                 const float* bl, char* ws,
                                                 float* out) {
  extern __shared__ char smem[];
  __bf16* vn = (__bf16*)smem;                 // [256][64]
  __bf16* vnT = (__bf16*)(smem + 32768);      // [64][256]
  __bf16* lvA = (__bf16*)(smem + 65536);      // [256][64]
  const float* hlast = (const float*)(ws + OFF_HLAST);
  const float* mean = (const float*)(ws + OFF_MEAN);
  const float* rstd = (const float*)(ws + OFF_RSTD);
  const char* wlc = ws + OFF_WL;
  int b = blockIdx.x;
  int o = threadIdx.x;
#pragma unroll
  for (int f = 0; f < DM_; ++f) {
    float x = hlast[((size_t)b * O_ + o) * DM_ + f];
    float y = (x - mean[f]) * rstd[f] * gamma[f] + beta[f];
    __bf16 yb = tobf(y);
    vn[o * DM_ + f] = yb;
    vnT[f * O_ + o] = yb;
  }
  __syncthreads();
  int lane = threadIdx.x & 31, widx = threadIdx.x >> 5;
  // phase 1: lv (256x64) = vn @ Wl^T + bl ; 64 tiles over 8 waves
  for (int idx = widx * 8; idx < widx * 8 + 8; ++idx) {
    int mt = idx >> 2, nt = idx & 3;
    float bv = bl[nt * 16 + (lane & 15)];
    v8f acc;
#pragma unroll
    for (int j = 0; j < 8; ++j) acc[j] = bv;
#pragma unroll
    for (int c = 0; c < 2; ++c) {
      v16bf a = *(const v16bf*)(vn + (mt * 16 + (lane & 15)) * DM_ + c * 32 + 16 * (lane >> 4));
      v16bf bb = ldfragG(wlc, c * 4 + nt, lane);
      acc = wmma_bf16(a, bb, acc);
    }
#pragma unroll
    for (int j = 0; j < 8; ++j) {
      int M = (lane < 16) ? j : j + 8;
      lvA[(mt * 16 + M) * DM_ + nt * 16 + (lane & 15)] = tobf(acc[j]);
    }
  }
  __syncthreads();
  // phase 2: out_b (256x256) = lv @ v^T ; 256 tiles over 8 waves
  for (int idx = widx * 32; idx < widx * 32 + 32; ++idx) {
    int mt = idx >> 4, nt = idx & 15;
    v8f acc = v8f_zero();
#pragma unroll
    for (int c = 0; c < 2; ++c) {
      v16bf a = *(const v16bf*)(lvA + (mt * 16 + (lane & 15)) * DM_ + c * 32 + 16 * (lane >> 4));
      v16bf bb = *(const v16bf*)(vn + (nt * 16 + (lane & 15)) * DM_ + c * 32 + 16 * (lane >> 4));
      acc = wmma_bf16(a, bb, acc);
    }
#pragma unroll
    for (int j = 0; j < 8; ++j) {
      int M = (lane < 16) ? j : j + 8;
      out[((size_t)b * O_ + mt * 16 + M) * O_ + nt * 16 + (lane & 15)] = acc[j];
    }
  }
}

// ---------------------------------------------------------------------------
extern "C" void kernel_launch(void* const* d_in, const int* in_sizes, int n_in,
                              void* d_out, int out_size, void* d_ws,
                              size_t ws_size, hipStream_t stream) {
  const float* X = (const float*)d_in[0];
  const float* dis = (const float*)d_in[1];
  const float* E = (const float*)d_in[2];
  const float* Wi = (const float*)d_in[3];
  const float* Wh = (const float*)d_in[4];
  const float* bi = (const float*)d_in[5];
  const float* bh = (const float*)d_in[6];
  const float* gamma = (const float*)d_in[7];
  const float* beta = (const float*)d_in[8];
  const float* Wl = (const float*)d_in[9];
  const float* bl = (const float*)d_in[10];
  char* ws = (char*)d_ws;
  float* out = (float*)d_out;

  k_zero<<<1, 128, 0, stream>>>(ws);
  k_geo<<<1, 256, 0, stream>>>(dis, E, (float*)(ws + OFF_GEO));
  k_prepw<<<8, 256, 0, stream>>>(Wi, Wh, Wl, ws);
  k_gtot<<<32, 256, 0, stream>>>(Wi, bi, bh, ws);
  k_flow<<<2048, 128, 4 * TDM_SLOT, stream>>>(X, E, ws);
  k_lstm<<<64, 256, 16384, stream>>>(ws);
  k_finalize<<<1, 64, 0, stream>>>(ws);
  k_head<<<32, 256, 98304, stream>>>(gamma, beta, bl, ws, out);
}
